// GPTGRNNDecoderSVar_17506286698872
// MI455X (gfx1250) — matrix operand order repaired and checked by reading, not compiled, not run on hardware
//
#include <hip/hip_runtime.h>
#include <hip/hip_bf16.h>
#include <math.h>

// ---------------- problem constants ----------------
constexpr int B    = 32;
constexpr int NMAX = 256;
constexpr int E    = 512;
constexpr int H    = 512;
constexpr int U    = 256;
constexpr int M    = 32;
constexpr int P    = 4;
constexpr int S    = NMAX + 1;          // 257
constexpr int IN0  = E + (M - 1) + P;   // 547
constexpr int XPAD = 576;               // padded layer-0 K (mult of 32)

typedef __attribute__((ext_vector_type(16))) _Float16 v16h;
typedef __attribute__((ext_vector_type(8)))  _Float16 v8h;
typedef __attribute__((ext_vector_type(8)))  float    v8f;

// ---------------------------------------------------------------------------
// WMMA GEMM:  C[Mrows x Ncols] = Ah[Mrows x K] * Wh[Ncols x K]^T (+ bias)
// f16 operands (pre-converted), f32 accumulate. One wave = one 16x16 C tile.
// Requirements (all call sites satisfy): Mrows,Ncols mult of 16; K mult of 32;
// lda,ldw mult of 8; Ah/Wh 16B-aligned.  Inner loop: 4x global_load_b128 + wmma.
// ---------------------------------------------------------------------------
__global__ __launch_bounds__(128)
void wmma_gemm_h(const _Float16* __restrict__ Ah, int lda,
                 const _Float16* __restrict__ Wh, int ldw,
                 const float* __restrict__ bias,
                 float* __restrict__ C, int ldc,
                 int Mrows, int Ncols, int K)
{
    const int wave   = threadIdx.x >> 5;
    const int lane   = threadIdx.x & 31;
    const int tileId = blockIdx.x * 4 + wave;
    const int tilesN = Ncols >> 4;
    const int tm     = tileId / tilesN;
    const int tn     = tileId - tm * tilesN;
    if (tm * 16 >= Mrows) return;                 // wave-uniform exit

    const int half = lane >> 4;                   // 0 | 1
    const int l15  = lane & 15;
    const _Float16* Ar = Ah + (size_t)(tm * 16 + l15) * lda;
    const _Float16* Wr = Wh + (size_t)(tn * 16 + l15) * ldw;

    v8f acc = {};
    for (int k0 = 0; k0 < K; k0 += 32) {
        // A fragment (16-bit A 16x32 ISA layout): per lane two contiguous
        // 8-element runs: K = k0+half*8 .. +7  and  K = k0+16+half*8 .. +7
        v8h alo = *(const v8h*)(Ar + k0 + half * 8);
        v8h ahi = *(const v8h*)(Ar + k0 + 16 + half * 8);
        // B fragment (32x16): per lane two contiguous 8-element runs:
        // K = k0+half*8 .. +7  and  K = k0+16+half*8 .. +7   (N = l15)
        v8h blo = *(const v8h*)(Wr + k0 + half * 8);
        v8h bhi = *(const v8h*)(Wr + k0 + 16 + half * 8);
        v16h a, b;
        #pragma unroll
        for (int j = 0; j < 8; ++j) {
            a[j] = alo[j]; a[8 + j] = ahi[j];
            b[j] = blo[j]; b[8 + j] = bhi[j];
        }
        acc = __builtin_amdgcn_wmma_f32_16x16x32_f16(
                  false, a, false, b, (short)0, acc, false, false);
    }
    // C/D layout: VGPR v -> row tm*16 + v + 8*half, col tn*16 + l15
    const int col = tn * 16 + l15;
    const float bb = bias ? bias[col] : 0.f;
    #pragma unroll
    for (int v = 0; v < 8; ++v) {
        int row = tm * 16 + v + 8 * half;
        C[(size_t)row * ldc + col] = acc[v] + bb;
    }
}

// fp32 [rows x Ksrc] -> f16 [rows x ldp], zero-padded K tail
__global__ void cvt_pad(const float* __restrict__ src, _Float16* __restrict__ dst,
                        int rows, int Ksrc, int ldp)
{
    int idx = blockIdx.x * blockDim.x + threadIdx.x;
    if (idx >= rows * ldp) return;
    int r = idx / ldp, k = idx - r * ldp;
    dst[idx] = (k < Ksrc) ? (_Float16)src[(size_t)r * Ksrc + k] : (_Float16)0.f;
}

// ---------------------------------------------------------------------------
// encoder_out = [emb_special[1] ; masked h_padded] -> d_out (fp32) + f16 mirror
// ---------------------------------------------------------------------------
__global__ void build_enc(const float* __restrict__ h_padded,
                          const float* __restrict__ emb_special,
                          const int*   __restrict__ num_nodes,
                          float* __restrict__ enc, _Float16* __restrict__ enc_h)
{
    size_t idx = (size_t)blockIdx.x * blockDim.x + threadIdx.x;
    if (idx >= (size_t)B * S * E) return;
    int e = idx % E;
    int s = (idx / E) % S;
    int b = idx / ((size_t)E * S);
    float v;
    if (s == 0) v = emb_special[E + e];                       // SOS embedding
    else {
        int j = s - 1;
        v = (j < num_nodes[b]) ? h_padded[((size_t)b * NMAX + j) * E + e] : 0.f;
    }
    enc[idx]   = v;
    enc_h[idx] = (_Float16)v;
}

// dec_hidden = tanh(hg @ Wp^T + bp) split into two layers (fp32 + f16 mirror)
__global__ void init_hidden(const float* __restrict__ tmp,
                            float* __restrict__ h0, float* __restrict__ h1,
                            _Float16* __restrict__ h0h, _Float16* __restrict__ h1h)
{
    int idx = blockIdx.x * blockDim.x + threadIdx.x;
    if (idx >= B * 2 * H) return;
    int b = idx >> 10, c = idx & 1023;
    float v = tanhf(tmp[idx]);
    if (c < H) { h0[b * H + c] = v;       h0h[b * H + c] = (_Float16)v; }
    else       { h1[b * H + c - H] = v;   h1h[b * H + c - H] = (_Float16)v; }
}

// zero coverage / loss partials; seed xbuf_h = [emb_special[0], 0, pe0, 0 pad]
__global__ void init_misc(const float* __restrict__ emb_special,
                          const float* __restrict__ emb_pos,
                          _Float16* __restrict__ xbuf_h,
                          float* __restrict__ cov, float* __restrict__ covp)
{
    int idx = blockIdx.x * blockDim.x + threadIdx.x;
    if (idx < B * S) cov[idx] = 0.f;
    if (idx < B)     covp[idx] = 0.f;
    if (idx < B * XPAD) {
        int c = idx % XPAD;
        float v;
        if      (c < E)           v = emb_special[c];          // emb_special[0]
        else if (c < E + M - 1)   v = 0.f;                     // theta0
        else if (c < IN0)         v = emb_pos[c - (E + M - 1)];// emb_pos[0]
        else                      v = 0.f;                     // K pad
        xbuf_h[idx] = (_Float16)v;
    }
}

// fused GRU gate combine: h = (1-z)*n + z*h  (fp32 master + f16 mirror)
__global__ void gru_update(const float* __restrict__ gi,
                           const float* __restrict__ gh,
                           float* __restrict__ h, _Float16* __restrict__ hh)
{
    int idx = blockIdx.x * blockDim.x + threadIdx.x;
    if (idx >= B * H) return;
    int b = idx / H, j = idx - b * H;
    const float* gib = gi + (size_t)b * 3 * H;
    const float* ghb = gh + (size_t)b * 3 * H;
    float r = 1.f / (1.f + expf(-(gib[j]     + ghb[j])));
    float z = 1.f / (1.f + expf(-(gib[H + j] + ghb[H + j])));
    float n = tanhf(gib[2 * H + j] + r * ghb[2 * H + j]);
    float v = (1.f - z) * n + z * h[idx];
    h[idx]  = v;
    hh[idx] = (_Float16)v;
}

// attention energies: uj[b,s] = sum_u V[u] * tanh(encW1 + w2out + cov*wc)
__global__ __launch_bounds__(128)
void att_kernel(const float* __restrict__ encW1, const float* __restrict__ w2,
                const float* __restrict__ cov,   const float* __restrict__ wc,
                const float* __restrict__ V,     float* __restrict__ uj)
{
    int gw   = (blockIdx.x * blockDim.x + threadIdx.x) >> 5;
    int lane = threadIdx.x & 31;
    if (gw >= B * S) return;
    int b = gw / S;
    float cv = cov[gw];
    const float* er = encW1 + (size_t)gw * U;
    const float* wb = w2 + (size_t)b * U;
    float p = 0.f;
    for (int u = lane; u < U; u += 32)
        p += V[u] * tanhf(er[u] + wb[u] + cv * wc[u]);
    #pragma unroll
    for (int o = 16; o; o >>= 1) p += __shfl_xor(p, o, 32);
    if (lane == 0) uj[gw] = p;
}

// per-batch-row: masked softmax, gumbel hard argmax, coverage update,
// chosen-row gather, theta head, next-step input assembly, output columns.
__global__ __launch_bounds__(256)
void fin_kernel(const float* __restrict__ uj,   const int* __restrict__ num_nodes,
                const float* __restrict__ u_noise,
                float* __restrict__ cov,        float* __restrict__ covp,
                const float* __restrict__ enc,  const float* __restrict__ h1,
                const float* __restrict__ Wf,   const float* __restrict__ bf,
                const float* __restrict__ emb_pos,
                _Float16* __restrict__ xbuf_h,
                float* __restrict__ out_ptr, float* __restrict__ out_adj,
                float* __restrict__ out_atts, int step)
{
    __shared__ float red[256];
    __shared__ int   redi[256];
    __shared__ float s_max, s_sum;
    __shared__ int   s_arg;
    const int b = blockIdx.x;
    const int t = threadIdx.x;
    const int len = num_nodes[b] + 1;

    // thread t owns s=t; thread 0 additionally owns s=256
    float u0 = uj[b * S + t];
    float m0 = (t < len) ? u0 : -1e30f;
    float m1 = -1e30f;
    if (t == 0) { float u1 = uj[b * S + 256]; m1 = (256 < len) ? u1 : -1e30f; }

    red[t] = fmaxf(m0, m1); __syncthreads();
    for (int o = 128; o; o >>= 1) { if (t < o) red[t] = fmaxf(red[t], red[t + o]); __syncthreads(); }
    if (t == 0) s_max = red[0]; __syncthreads();
    float mx = s_max;

    float e0 = expf(m0 - mx);
    float e1 = (t == 0) ? expf(m1 - mx) : 0.f;
    red[t] = e0 + e1; __syncthreads();
    for (int o = 128; o; o >>= 1) { if (t < o) red[t] += red[t + o]; __syncthreads(); }
    if (t == 0) s_sum = red[0]; __syncthreads();
    float inv = 1.0f / s_sum;
    float a0 = e0 * inv;
    float a1 = e1 * inv;                         // valid on t==0 only

    // gumbel scores (TAU=1: softmax/div don't change argmax)
    const float* un = u_noise + ((size_t)step * B + b) * S;
    float g0 = -logf(-logf(un[t] + 1e-10f) + 1e-10f);
    float sc = logf(a0 + 1e-12f) + g0;
    int   ix = t;
    if (t == 0) {
        float g1  = -logf(-logf(un[256] + 1e-10f) + 1e-10f);
        float sc1 = logf(a1 + 1e-12f) + g1;
        if (sc1 > sc) { sc = sc1; ix = 256; }    // strict > keeps lowest index
    }
    red[t] = sc; redi[t] = ix; __syncthreads();
    for (int o = 128; o; o >>= 1) {
        if (t < o) {
            if (red[t + o] > red[t] ||
               (red[t + o] == red[t] && redi[t + o] < redi[t])) {
                red[t] = red[t + o]; redi[t] = redi[t + o];
            }
        }
        __syncthreads();
    }
    if (t == 0) s_arg = redi[0]; __syncthreads();
    const int amax = s_arg;

    // coverage loss / update, atts & pointer output columns
    float cv0 = cov[b * S + t];
    float cl  = fminf(a0, cv0);
    cov[b * S + t] = cv0 + a0;
    out_atts[((size_t)b * S + t) * M + step] = a0;
    out_ptr [((size_t)b * S + t) * M + step] = (t == amax) ? 1.f : 0.f;
    if (t == 0) {
        float cv1 = cov[b * S + 256];
        cl += fminf(a1, cv1);
        cov[b * S + 256] = cv1 + a1;
        out_atts[((size_t)b * S + 256) * M + step] = a1;
        out_ptr [((size_t)b * S + 256) * M + step] = (256 == amax) ? 1.f : 0.f;
    }
    red[t] = cl; __syncthreads();
    for (int o = 128; o; o >>= 1) { if (t < o) red[t] += red[t + o]; __syncthreads(); }
    if (t == 0) covp[b] += red[0];               // single writer -> deterministic

    // chosen = encoder_out[b, amax, :] -> f16 next-step input
    for (int c = t; c < E; c += 256)
        xbuf_h[b * XPAD + c] = (_Float16)enc[((size_t)b * S + amax) * E + c];
    // theta = sigmoid(h1 @ Wf^T + bf) (zeros at step 0)
    if (t < M - 1) {
        float acc = bf[t];
        const float* hr = h1 + (size_t)b * H;
        const float* wr = Wf + (size_t)t * H;
        for (int k = 0; k < H; ++k) acc += hr[k] * wr[k];
        float th = (step < 1) ? 0.f : 1.f / (1.f + expf(-acc));
        xbuf_h[b * XPAD + E + t] = (_Float16)th;
        out_adj[((size_t)b * (M - 1) + t) * M + step] = th;
    }
    // position embedding for next step
    if (t < P && (step + 1) < M)
        xbuf_h[b * XPAD + E + (M - 1) + t] = (_Float16)emb_pos[(step + 1) * P + t];
}

__global__ void finalize(const float* __restrict__ covp, float* __restrict__ out)
{
    __shared__ float r[32];
    int t = threadIdx.x;
    r[t] = covp[t]; __syncthreads();
    for (int o = 16; o; o >>= 1) { if (t < o) r[t] += r[t + o]; __syncthreads(); }
    if (t == 0) out[0] = r[0] / (float)B;
}

// ---------------------------------------------------------------------------
static inline int gemm_blocks(int Mr, int Nc) { return ((Mr / 16) * (Nc / 16) + 3) / 4; }
static inline int grid1(int n) { return (n + 255) / 256; }

extern "C" void kernel_launch(void* const* d_in, const int* in_sizes, int n_in,
                              void* d_out, int out_size, void* d_ws, size_t ws_size,
                              hipStream_t stream)
{
    (void)in_sizes; (void)n_in; (void)out_size; (void)ws_size;

    const float* h_padded    = (const float*)d_in[0];
    const float* hg          = (const float*)d_in[1];
    const int*   num_nodes   = (const int*)  d_in[2];
    const float* u_noise     = (const float*)d_in[3];
    const float* emb_special = (const float*)d_in[4];
    const float* emb_pos     = (const float*)d_in[5];
    const float* Wp    = (const float*)d_in[6];
    const float* bp    = (const float*)d_in[7];
    const float* W_ih0 = (const float*)d_in[8];
    const float* W_hh0 = (const float*)d_in[9];
    const float* b_ih0 = (const float*)d_in[10];
    const float* b_hh0 = (const float*)d_in[11];
    const float* W_ih1 = (const float*)d_in[12];
    const float* W_hh1 = (const float*)d_in[13];
    const float* b_ih1 = (const float*)d_in[14];
    const float* b_hh1 = (const float*)d_in[15];
    const float* W1    = (const float*)d_in[16];
    const float* W2    = (const float*)d_in[17];
    const float* wc    = (const float*)d_in[18];
    const float* V     = (const float*)d_in[19];
    const float* Wf    = (const float*)d_in[20];
    const float* bf    = (const float*)d_in[21];

    // output layout (floats): pointer | cov_loss | encoder_out | adj | atts
    float* out         = (float*)d_out;
    float* out_ptr     = out;                                // B*S*M
    float* out_covloss = out + (size_t)B * S * M;            // 1
    float* enc         = out_covloss + 1;                    // B*S*E (fp32, gathered from)
    float* out_adj     = enc + (size_t)B * S * E;            // B*(M-1)*M
    float* out_atts    = out_adj + (size_t)B * (M - 1) * M;  // B*S*M

    // ---- workspace: f16 region first (16B aligned), fp32 region after ----
    _Float16* h16     = (_Float16*)d_ws;
    _Float16* enc_h   = h16;                                 // B*S*E     = 4210688
    _Float16* w_ih0_h = enc_h   + (size_t)B * S * E;         // 1536*576  = 884736
    _Float16* w_hh0_h = w_ih0_h + (size_t)(3 * H) * XPAD;    // 1536*512  = 786432
    _Float16* w_ih1_h = w_hh0_h + (size_t)(3 * H) * H;
    _Float16* w_hh1_h = w_ih1_h + (size_t)(3 * H) * H;
    _Float16* w2_h    = w_hh1_h + (size_t)(3 * H) * H;       // 256*512
    _Float16* w1_h    = w2_h    + (size_t)U * H;             // 256*512
    _Float16* wp_h    = w1_h    + (size_t)U * E;             // 1024*512
    _Float16* xbuf_h  = wp_h    + (size_t)(2 * H) * E;       // B*XPAD = 18432
    _Float16* h0_h    = xbuf_h  + B * XPAD;                  // 16384
    _Float16* h1_h    = h0_h    + B * H;                     // 16384
    _Float16* hg_h    = h1_h    + B * H;                     // 16384
    _Float16* h16_end = hg_h    + B * E;

    float* wsf   = (float*)h16_end;                          // 16B aligned (even # halves)
    float* encW1 = wsf;                                      // B*S*U = 2105344
    float* tmp   = encW1 + (size_t)B * S * U;                // B*2H
    float* h0    = tmp   + B * 2 * H;
    float* h1    = h0    + B * H;
    float* gi    = h1    + B * H;                            // B*3H
    float* gh    = gi    + B * 3 * H;
    float* w2    = gh    + B * 3 * H;                        // B*U
    float* uj    = w2    + B * U;                            // B*S
    float* cov   = uj    + B * S;                            // B*S
    float* covp  = cov   + B * S;                            // B

    // ---- weight / activation conversions (once per call) ----
    cvt_pad<<<grid1(3 * H * XPAD), 256, 0, stream>>>(W_ih0, w_ih0_h, 3 * H, IN0, XPAD);
    cvt_pad<<<grid1(3 * H * H),    256, 0, stream>>>(W_hh0, w_hh0_h, 3 * H, H, H);
    cvt_pad<<<grid1(3 * H * H),    256, 0, stream>>>(W_ih1, w_ih1_h, 3 * H, H, H);
    cvt_pad<<<grid1(3 * H * H),    256, 0, stream>>>(W_hh1, w_hh1_h, 3 * H, H, H);
    cvt_pad<<<grid1(U * H),        256, 0, stream>>>(W2, w2_h, U, H, H);
    cvt_pad<<<grid1(U * E),        256, 0, stream>>>(W1, w1_h, U, E, E);
    cvt_pad<<<grid1(2 * H * E),    256, 0, stream>>>(Wp, wp_h, 2 * H, E, E);
    cvt_pad<<<grid1(B * E),        256, 0, stream>>>(hg, hg_h, B, E, E);

    // ---- loop-invariant work ----
    build_enc<<<grid1(B * S * E), 256, 0, stream>>>(h_padded, emb_special, num_nodes, enc, enc_h);
    wmma_gemm_h<<<gemm_blocks(B * S, U), 128, 0, stream>>>(
        enc_h, E, w1_h, E, nullptr, encW1, U, B * S, U, E);
    wmma_gemm_h<<<gemm_blocks(B, 2 * H), 128, 0, stream>>>(
        hg_h, E, wp_h, E, bp, tmp, 2 * H, B, 2 * H, E);
    init_hidden<<<grid1(B * 2 * H), 256, 0, stream>>>(tmp, h0, h1, h0_h, h1_h);
    init_misc<<<grid1(B * XPAD), 256, 0, stream>>>(emb_special, emb_pos, xbuf_h, cov, covp);

    // ---- M sequential decode steps ----
    for (int i = 0; i < M; ++i) {
        // GRU layer 0
        wmma_gemm_h<<<gemm_blocks(B, 3 * H), 128, 0, stream>>>(
            xbuf_h, XPAD, w_ih0_h, XPAD, b_ih0, gi, 3 * H, B, 3 * H, XPAD);
        wmma_gemm_h<<<gemm_blocks(B, 3 * H), 128, 0, stream>>>(
            h0_h, H, w_hh0_h, H, b_hh0, gh, 3 * H, B, 3 * H, H);
        gru_update<<<grid1(B * H), 256, 0, stream>>>(gi, gh, h0, h0_h);
        // GRU layer 1
        wmma_gemm_h<<<gemm_blocks(B, 3 * H), 128, 0, stream>>>(
            h0_h, H, w_ih1_h, H, b_ih1, gi, 3 * H, B, 3 * H, H);
        wmma_gemm_h<<<gemm_blocks(B, 3 * H), 128, 0, stream>>>(
            h1_h, H, w_hh1_h, H, b_hh1, gh, 3 * H, B, 3 * H, H);
        gru_update<<<grid1(B * H), 256, 0, stream>>>(gi, gh, h1, h1_h);
        // attention query projection
        wmma_gemm_h<<<gemm_blocks(B, U), 128, 0, stream>>>(
            h1_h, H, w2_h, H, nullptr, w2, U, B, U, H);
        // attention energies + softmax/gumbel/coverage/next-input
        att_kernel<<<(B * S + 3) / 4, 128, 0, stream>>>(encW1, w2, cov, wc, V, uj);
        fin_kernel<<<B, 256, 0, stream>>>(uj, num_nodes, u_noise, cov, covp, enc, h1,
                                          Wf, bf, emb_pos, xbuf_h,
                                          out_ptr, out_adj, out_atts, i);
    }
    finalize<<<1, 32, 0, stream>>>(covp, out_covloss);
}
